// ReLUKANLayer_35424890257937
// MI455X (gfx1250) — compile-verified, matrix operand order checked by reading, not run on hardware
//
#include <hip/hip_runtime.h>
#include <hip/hip_bf16.h>
#include <cstdint>
#include <cstring>

// ---- problem constants (from the reference) ----
#define B_ROWS   8192
#define IN_DIM   1024
#define OUT_DIM  512
#define G_DIM    20
#define NPASS    (G_DIM + 1)          // 1 base (silu) pass + 20 spline (relu) passes
#define KTOT     (NPASS * IN_DIM)     // 21504
#define BM       128
#define BN       128

typedef __attribute__((ext_vector_type(16))) __bf16 v16bf;
typedef __attribute__((ext_vector_type(8)))  float  v8f;

union BF16x16 {
    v16bf    v;
    uint32_t u[8];
};

// ---------------------------------------------------------------------------
// Weight repack: f32 (base_weight, spline_weight*scale) -> bf16 B-fragments in
// the native WMMA 16-bit 32x16 per-lane layout. One thread per output half,
// stores fully coalesced; gathered f32 loads are serviced by L2.
//
// flat index = ((kt*32 + nt)*32 + lane)*16 + j
//   n   = nt*16 + (lane & 15)
//   kin = (lane>>4)*8 + (j & 7) + (j>=8 ? 16 : 0)     (K within the 32-tile)
//   kglob = kt*32 + kin ;  pass = kglob/1024 ; i = kglob%1024
// ---------------------------------------------------------------------------
__global__ __launch_bounds__(256)
void prep_weights(const float* __restrict__ bw,    // (OUT, IN)
                  const float* __restrict__ sw,    // (OUT, IN, G)
                  const float* __restrict__ scale, // (OUT)
                  __bf16* __restrict__ frag)
{
    const int tid = blockIdx.x * blockDim.x + threadIdx.x; // < KTOT*OUT, exact multiple of 256
    const int j    = tid & 15;
    const int lane = (tid >> 4) & 31;
    const int nt   = (tid >> 9) & 31;     // 32 N-tiles of 16
    const int kt   = tid >> 14;           // 672 K-tiles of 32

    const int n   = nt * 16 + (lane & 15);
    const int kin = ((lane >> 4) << 3) + (j & 7) + ((j >> 3) << 4);
    const int kglob = kt * 32 + kin;
    const int pass  = kglob >> 10;        // / IN_DIM
    const int i     = kglob & (IN_DIM - 1);

    float v;
    if (pass == 0) {
        v = bw[(size_t)n * IN_DIM + i];
    } else {
        v = sw[((size_t)n * IN_DIM + i) * G_DIM + (pass - 1)] * scale[n];
    }
    frag[tid] = (__bf16)v;
}

// ---------------------------------------------------------------------------
// Fused KAN GEMM. Loop order: outer = 32 K-chunks of x (loaded + converted to
// bf16 ONCE), inner = 21 activation passes (f32 silu + 20 packed-bf16 relu
// shifts via explicit v_pk_* inline asm), each doing 8
// v_wmma_f32_16x16x32_bf16 against pre-packed B-fragments.
// A 16x32 bf16 fragment layout (ISA 7.12.2): lane 0-15 -> M, K 0..7 & 16..23;
// lane 16-31 -> same M, K 8..15 & 24..31.
// ---------------------------------------------------------------------------
__device__ __forceinline__ float silu_f(float v) {
    // v * sigmoid(v); fast v_rcp_f32 instead of IEEE divide expansion
    return v * __builtin_amdgcn_rcpf(1.0f + __expf(-v));
}

// relu(x - g) on a packed bf16 pair: ng2 = (-g, -g) as bf16x2 bits
__device__ __forceinline__ uint32_t pk_relu_shift(uint32_t x2, uint32_t ng2) {
    uint32_t t;
    asm("v_pk_add_bf16 %0, %1, %2" : "=v"(t) : "v"(x2), "v"(ng2));
    asm("v_pk_max_num_bf16 %0, %1, 0" : "=v"(t) : "v"(t));
    return t;
}

// load the 4 B-fragments and issue the 8 WMMAs for one K-tile
__device__ __forceinline__ void mma_step(const v16bf afrag[2],
                                         const __bf16* __restrict__ frag,
                                         int kt, int ntBase, int lane,
                                         v8f acc[2][4])
{
    v16bf bfrag[4];
#pragma unroll
    for (int nt = 0; nt < 4; ++nt) {
        const __bf16* bp =
            frag + (((size_t)(kt * 32 + ntBase + nt) * 32 + lane) << 4);
        bfrag[nt] = *(const v16bf*)bp;   // 32 contiguous bytes per lane
    }
#pragma unroll
    for (int mt = 0; mt < 2; ++mt)
#pragma unroll
        for (int nt = 0; nt < 4; ++nt)
            acc[mt][nt] = __builtin_amdgcn_wmma_f32_16x16x32_bf16(
                false, afrag[mt], false, bfrag[nt],
                (short)0, acc[mt][nt], false, false);
}

__global__ __launch_bounds__(256)
void kan_gemm(const float* __restrict__ x,     // (B, IN)
              const float* __restrict__ grid,  // (G)
              const __bf16* __restrict__ frag, // packed weights (KTOT x OUT)
              float* __restrict__ out)         // (B, OUT)
{
    const int lane = threadIdx.x & 31;
    const int wave = threadIdx.x >> 5;   // 8 waves
    const int wm   = wave >> 1;          // 0..3 (M)
    const int wn   = wave & 1;           // 0..1 (N)
    const int blockM = blockIdx.x * BM;
    const int blockN = blockIdx.y * BN;
    const int mlow  = lane & 15;
    const int klgrp = (lane >> 4) << 3;  // 0 or 8

    v8f acc[2][4];
#pragma unroll
    for (int mt = 0; mt < 2; ++mt)
#pragma unroll
        for (int nt = 0; nt < 4; ++nt)
            acc[mt][nt] = (v8f){0.f, 0.f, 0.f, 0.f, 0.f, 0.f, 0.f, 0.f};

    const float* xr[2];
    xr[0] = x + (size_t)(blockM + wm * 32 + 0  + mlow) * IN_DIM;
    xr[1] = x + (size_t)(blockM + wm * 32 + 16 + mlow) * IN_DIM;
    const int ntBase = (blockN + wn * 64) >> 4;

    // preload grid knots, negated, as duplicated bf16 pairs (uniform)
    uint32_t ngb[G_DIM];
#pragma unroll
    for (int g = 0; g < G_DIM; ++g) {
        __bf16 b = (__bf16)(-grid[g]);
        uint16_t bits;
        __builtin_memcpy(&bits, &b, 2);
        ngb[g] = ((uint32_t)bits << 16) | bits;
    }

    for (int kk = 0; kk < IN_DIM; kk += 32) {
        const int kkIdx = kk >> 5;

        // ---- load raw x fragment floats ONCE per K-chunk ----
        float f[2][16];
#pragma unroll
        for (int mt = 0; mt < 2; ++mt) {
            const float* p = xr[mt] + kk + klgrp;
            const float4 a0 = *(const float4*)(p + 0);
            const float4 a1 = *(const float4*)(p + 4);
            const float4 b0 = *(const float4*)(p + 16);
            const float4 b1 = *(const float4*)(p + 20);
            f[mt][0]=a0.x; f[mt][1]=a0.y; f[mt][2]=a0.z; f[mt][3]=a0.w;
            f[mt][4]=a1.x; f[mt][5]=a1.y; f[mt][6]=a1.z; f[mt][7]=a1.w;
            f[mt][8]=b0.x; f[mt][9]=b0.y; f[mt][10]=b0.z; f[mt][11]=b0.w;
            f[mt][12]=b1.x; f[mt][13]=b1.y; f[mt][14]=b1.z; f[mt][15]=b1.w;
        }

        // ---- pass 0: silu(x) @ base_weight^T (f32 fast path) ----
        {
            v16bf afrag[2];
#pragma unroll
            for (int mt = 0; mt < 2; ++mt)
#pragma unroll
                for (int jj = 0; jj < 16; ++jj)
                    afrag[mt][jj] = (__bf16)silu_f(f[mt][jj]);
            mma_step(afrag, frag, kkIdx, ntBase, lane, acc);
        }

        // ---- convert x to packed bf16 ONCE for the 20 relu passes ----
        BF16x16 xb[2];
#pragma unroll
        for (int mt = 0; mt < 2; ++mt)
#pragma unroll
            for (int jj = 0; jj < 16; ++jj)
                xb[mt].v[jj] = (__bf16)f[mt][jj];   // -> v_cvt_pk_bf16_f32

        // ---- passes 1..20: relu(x - grid[g]) via v_pk_add/_max_num_bf16 ----
#pragma unroll 4
        for (int g = 0; g < G_DIM; ++g) {
            const uint32_t ng2 = ngb[g];
            v16bf afrag[2];
#pragma unroll
            for (int mt = 0; mt < 2; ++mt) {
                BF16x16 a;
#pragma unroll
                for (int r = 0; r < 8; ++r)
                    a.u[r] = pk_relu_shift(xb[mt].u[r], ng2);
                afrag[mt] = a.v;
            }
            mma_step(afrag, frag, (g + 1) * 32 + kkIdx, ntBase, lane, acc);
        }
    }

    // ---- store C (f32): VGPR v -> M, lane&15 -> N, lane>=16 -> M+8 ----
#pragma unroll
    for (int mt = 0; mt < 2; ++mt) {
#pragma unroll
        for (int nt = 0; nt < 4; ++nt) {
            const int col     = blockN + wn * 64 + nt * 16 + mlow;
            const int rowBase = blockM + wm * 32 + mt * 16 + klgrp;
#pragma unroll
            for (int v = 0; v < 8; ++v)
                out[(size_t)(rowBase + v) * OUT_DIM + col] = acc[mt][nt][v];
        }
    }
}

// ---------------------------------------------------------------------------
extern "C" void kernel_launch(void* const* d_in, const int* in_sizes, int n_in,
                              void* d_out, int out_size, void* d_ws, size_t ws_size,
                              hipStream_t stream)
{
    const float* x     = (const float*)d_in[0]; // (8192, 1024)
    const float* bw    = (const float*)d_in[1]; // (512, 1024)
    const float* sw    = (const float*)d_in[2]; // (512, 1024, 20)
    const float* scale = (const float*)d_in[3]; // (512, 1, 1)
    const float* grid  = (const float*)d_in[4]; // (20)
    float* out = (float*)d_out;

    __bf16* frag = (__bf16*)d_ws;               // KTOT*OUT halfs = 22 MB

    {
        const int total   = KTOT * OUT_DIM;     // 11,010,048 (multiple of 256)
        const int threads = 256;
        prep_weights<<<total / threads, threads, 0, stream>>>(bw, sw, scale, frag);
    }
    {
        dim3 g(B_ROWS / BM, OUT_DIM / BN);      // 64 x 4
        kan_gemm<<<g, 256, 0, stream>>>(x, grid, frag, out);
    }
}
